// PTSDGraphNet_62904091017701
// MI455X (gfx1250) — compile-verified, matrix operand order checked by reading, not run on hardware
//
#include <hip/hip_runtime.h>
#include <math.h>

// ---------------- constants ----------------
constexpr int   kN   = 8192;
constexpr int   kE   = 262144;
constexpr int   kG   = 16;
constexpr int   kD   = 512;
constexpr int   kP   = 128;
constexpr int   kL   = 3;
constexpr int   kRBF = 48;
constexpr float kCUT = 5.0f;
constexpr float kGAMMA = 46.08f;               // 0.5*(48/5)^2
constexpr float kSTEP  = 5.0f / 47.0f;         // linspace(0,5,48) step
constexpr float kC1    = 0.48860251190291992f; // sqrt(3/(4pi))
constexpr float kPIf   = 3.14159265358979323846f;

typedef __attribute__((ext_vector_type(16))) __bf16 v16bf;
typedef __attribute__((ext_vector_type(8)))  float  v8f;
typedef __attribute__((ext_vector_type(2)))  __bf16 v2bf;

// Async global->LDS copy path (CDNA5): guarded so the build stays green if the
// toolchain doesn't expose the builtins; falls back to plain loads.
#if defined(__has_builtin)
#  if __has_builtin(__builtin_amdgcn_global_load_async_to_lds_b64)
#    define USE_ASYNC_LDS 1
#  endif
#  if __has_builtin(__builtin_amdgcn_cvt_pk_bf16_f32)
#    define USE_CVT_PK_BF16 1
#  endif
#endif

#ifdef USE_ASYNC_LDS
// builtin signature: (global v2i32*, local v2i32*, imm int offset, imm int cpol)
typedef int v2i32 __attribute__((vector_size(8)));
typedef __attribute__((address_space(1))) v2i32 as1_v2i32;
typedef __attribute__((address_space(3))) v2i32 as3_v2i32;
#endif

__device__ __forceinline__ void wait_async_then_barrier() {
#if defined(__has_builtin)
#  if __has_builtin(__builtin_amdgcn_s_wait_asynccnt)
    __builtin_amdgcn_s_wait_asynccnt(0);
#  else
    asm volatile("s_wait_asynccnt 0" ::: "memory");
#  endif
#else
    asm volatile("s_wait_asynccnt 0" ::: "memory");
#endif
    __syncthreads();
}

__device__ __forceinline__ unsigned short f2bf(float f) {
    union { float f; unsigned u; } x; x.f = f;
    unsigned u = x.u;
    u += 0x7FFFu + ((u >> 16) & 1u);
    return (unsigned short)(u >> 16);
}
// pack two f32 into packed bf16 (RNE); hardware v_cvt_pk_bf16_f32 when available
__device__ __forceinline__ unsigned pack2bf(float a, float b) {
#ifdef USE_CVT_PK_BF16
    union { v2bf v; unsigned u; } r;
    r.v = __builtin_amdgcn_cvt_pk_bf16_f32(a, b);
    return r.u;
#else
    return (unsigned)f2bf(a) | ((unsigned)f2bf(b) << 16);
#endif
}
__device__ __forceinline__ float sigm(float a) { return 1.0f / (1.0f + __expf(-a)); }
__device__ __forceinline__ float silu(float a) { return a * sigm(a); }
__device__ __forceinline__ float dsilu(float a) {
    float s = sigm(a);
    return s * (1.0f + a * (1.0f - s));
}

// ---------------- generic bf16 WMMA GEMM: C[M,N] = A[M,K] @ B + bias ----------------
// A row-major f32 (lda). B element (k,n) = Bp[k*bsk + n*bsn] (handles B and B^T).
// REQUIRES: M % 128 == 0, N % 64 == 0, K % 32 == 0 (true for every call here),
// and either bsn==1 (row-major B) or bsk==1 (transposed B).
#define BM 128
#define BN 64
#define BK 32
__global__ __launch_bounds__(256)
void gemm_bf16_k(const float* __restrict__ A, int lda,
                 const float* __restrict__ Bp, long bsk, long bsn,
                 const float* __restrict__ bias,
                 float* __restrict__ C, int ldc,
                 int M, int K, int Nn)
{
    __shared__ __align__(16) unsigned short Al[BM * BK]; // [m][k]
    __shared__ __align__(16) unsigned short Bl[BN * BK]; // [n][k]
    const int m0 = blockIdx.x * BM, n0 = blockIdx.y * BN;
    const int tid = threadIdx.x, lane = tid & 31, wave = tid >> 5;
    union Acc { v8f v; float f[8]; } acc[4];
#pragma unroll
    for (int c = 0; c < 4; ++c)
#pragma unroll
        for (int j = 0; j < 8; ++j) acc[c].f[j] = 0.0f;

    for (int k0 = 0; k0 < K; k0 += BK) {
        // ---- stage A tile 128x32: issue all 4 b128 loads, then convert+store ----
        float4 va[4];
#pragma unroll
        for (int q = 0; q < 4; ++q) {
            int id = tid + 256 * q;
            int rm = id >> 3, rk4 = (id & 7) << 2;
            va[q] = *(const float4*)(A + (long)(m0 + rm) * lda + k0 + rk4);
        }
        // ---- stage B tile into Bl[n][k]: issue loads, then convert+store ----
        float4 vb[2];
        if (bsn == 1) { // row-major B [K,N]: contiguous along n
#pragma unroll
            for (int q = 0; q < 2; ++q) {
                int id = tid + 256 * q;            // 512 slots
                int rk = id >> 4, rn4 = (id & 15) << 2;
                vb[q] = *(const float4*)(Bp + (long)(k0 + rk) * bsk + (n0 + rn4));
            }
#pragma unroll
            for (int q = 0; q < 2; ++q) {
                int id = tid + 256 * q;
                int rk = id >> 4, rn4 = (id & 15) << 2;
                Bl[(rn4 + 0) * BK + rk] = f2bf(vb[q].x);
                Bl[(rn4 + 1) * BK + rk] = f2bf(vb[q].y);
                Bl[(rn4 + 2) * BK + rk] = f2bf(vb[q].z);
                Bl[(rn4 + 3) * BK + rk] = f2bf(vb[q].w);
            }
        } else {        // transposed B (bsk==1): contiguous along k
#pragma unroll
            for (int q = 0; q < 2; ++q) {
                int id = tid + 256 * q;
                int rn = id >> 3, rk4 = (id & 7) << 2;
                vb[q] = *(const float4*)(Bp + (long)(n0 + rn) * bsn + (k0 + rk4));
            }
#pragma unroll
            for (int q = 0; q < 2; ++q) {
                int id = tid + 256 * q;
                int rn = id >> 3, rk4 = (id & 7) << 2;
                *(uint2*)&Bl[rn * BK + rk4] =
                    make_uint2(pack2bf(vb[q].x, vb[q].y), pack2bf(vb[q].z, vb[q].w));
            }
        }
#pragma unroll
        for (int q = 0; q < 4; ++q) {
            int id = tid + 256 * q;
            int rm = id >> 3, rk4 = (id & 7) << 2;
            *(uint2*)&Al[rm * BK + rk4] =
                make_uint2(pack2bf(va[q].x, va[q].y), pack2bf(va[q].z, va[q].w));
        }
        __syncthreads();
        const int hs = (lane >> 4) & 1;
        union Fr { v16bf v; unsigned d[8]; } af, bfr;
        const unsigned* Ar = (const unsigned*)&Al[(wave * 16 + (lane & 15)) * BK];
#pragma unroll
        for (int t = 0; t < 4; ++t) { af.d[t] = Ar[hs * 4 + t]; af.d[4 + t] = Ar[8 + hs * 4 + t]; }
#pragma unroll
        for (int c = 0; c < 4; ++c) {
            const unsigned* Br = (const unsigned*)&Bl[((c << 4) + (lane & 15)) * BK];
#pragma unroll
            for (int t = 0; t < 4; ++t) { bfr.d[t] = Br[hs * 4 + t]; bfr.d[4 + t] = Br[8 + hs * 4 + t]; }
            acc[c].v = __builtin_amdgcn_wmma_f32_16x16x32_bf16(
                false, af.v, false, bfr.v, (short)0, acc[c].v, false, false);
        }
        __syncthreads();
    }
    const int hs = (lane >> 4) & 1;
#pragma unroll
    for (int c = 0; c < 4; ++c) {
        int col = n0 + c * 16 + (lane & 15);
        float bv = bias ? bias[col] : 0.0f;
#pragma unroll
        for (int j = 0; j < 8; ++j) {
            int row = m0 + wave * 16 + hs * 8 + j;
            C[(long)row * ldc + col] = acc[c].f[j] + bv;
        }
    }
}

// ---------------- small elementwise / utility kernels ----------------
__global__ void zero_k(float* p, long n) {
    long i = (long)blockIdx.x * blockDim.x + threadIdx.x;
    if (i < n) p[i] = 0.0f;
}
__global__ void silu_k(float* dst, const float* src, long n) {
    long i = (long)blockIdx.x * blockDim.x + threadIdx.x;
    if (i < n) dst[i] = silu(src[i]);
}
__global__ void muldsilu_k(float* dst, const float* a, long n) {
    long i = (long)blockIdx.x * blockDim.x + threadIdx.x;
    if (i < n) dst[i] *= dsilu(a[i]);
}
__global__ void add3_k(float* dst, const float* a, const float* b, long n) {
    long i = (long)blockIdx.x * blockDim.x + threadIdx.x;
    if (i < n) dst[i] = a[i] + b[i];
}
__global__ void addip_k(float* dst, const float* src, long n) {
    long i = (long)blockIdx.x * blockDim.x + threadIdx.x;
    if (i < n) dst[i] += src[i];
}
__global__ void embed_k(float* x0, const float* z_emb, const int* z) {
    long i = (long)blockIdx.x * blockDim.x + threadIdx.x;
    if (i >= (long)kN * kD) return;
    int n = (int)(i >> 9), d = (int)(i & 511);
    x0[i] = z_emb[(long)(z[n] - 1) * kD + d];
}
__global__ void pos_k(float* pos, const float* frac, const float* cell, const int* batch) {
    int n = blockIdx.x * blockDim.x + threadIdx.x;
    if (n >= kN) return;
    int g = batch[n];
    for (int k = 0; k < 3; ++k) {
        float s = 0.0f;
        for (int d = 0; d < 3; ++d) s += frac[n * 3 + d] * cell[g * 9 + d * 3 + k];
        pos[n * 3 + k] = s;
    }
}
__global__ void edge_geom_k(const float* pos, const float* cell, const float* image,
                            const int* ej, const int* ei, const int* bj,
                            float* vr, float* rr, float* fac) {
    int e = blockIdx.x * blockDim.x + threadIdx.x;
    if (e >= kE) return;
    int j = ej[e], i = ei[e], g = bj[e];
    float v[3], s2 = 0.0f;
    for (int k = 0; k < 3; ++k) {
        float sh = image[e * 3 + 0] * cell[g * 9 + 0 + k] +
                   image[e * 3 + 1] * cell[g * 9 + 3 + k] +
                   image[e * 3 + 2] * cell[g * 9 + 6 + k];
        v[k] = pos[j * 3 + k] + sh - pos[i * 3 + k];
        s2 += v[k] * v[k];
        vr[(long)e * 3 + k] = v[k];
    }
    float r = sqrtf(s2 + 1e-12f);
    rr[e] = r;
    fac[e] = (r < kCUT) ? 0.5f * (__cosf(kPIf * r / kCUT) + 1.0f) : 0.0f;
}

// Stage the per-layer rbf_W slice [48 x 128] into LDS (async-to-LDS when available).
__device__ __forceinline__ void stage_Wl(float* Wl, const float* __restrict__ rbfW,
                                         int l, int tid) {
#ifdef USE_ASYNC_LDS
    // rows are 128 contiguous floats in both src and dst -> b64 (2-float) chunks
    for (int idx = tid * 2; idx < kRBF * kP; idx += 256 * 2) {
        int k = idx >> 7, p = idx & 127;
        __builtin_amdgcn_global_load_async_to_lds_b64(
            (as1_v2i32*)(rbfW + (long)k * (kL * kP) + l * kP + p),
            (as3_v2i32*)&Wl[idx], 0, 0);
    }
#else
    for (int idx = tid; idx < kRBF * kP; idx += 256) {
        int k = idx >> 7, p = idx & 127;
        Wl[idx] = rbfW[(long)k * (kL * kP) + l * kP + p];
    }
#endif
}

// ---------------- fused edge forward (gather -> rbf mix -> scatter) ----------------
__global__ __launch_bounds__(256)
void edge_fwd_k(const int* __restrict__ ej, const int* __restrict__ ei,
                const float* __restrict__ hbuf,
                const float* __restrict__ rbfW, const float* __restrict__ rbfB, int l,
                const float* __restrict__ vr, const float* __restrict__ rr,
                const float* __restrict__ fac,
                float* __restrict__ aggcat, float* __restrict__ m_out)
{
    __shared__ float Wl[kRBF * kP];
    __shared__ float gsh[2][kRBF];
    const int tid = threadIdx.x;
    stage_Wl(Wl, rbfW, l, tid);
    const int sub = tid >> 7, p = tid & 127;
    const long e = (long)blockIdx.x * 2 + sub;
    const bool ok = (e < kE);
    float r = 1.0f, f = 0.0f; int j = 0, i = 0;
    if (ok) { r = rr[e]; f = fac[e]; j = ej[e]; i = ei[e]; }
    if (ok && p < kRBF) {
        float d = r - kSTEP * (float)p;
        gsh[sub][p] = __expf(-kGAMMA * d * d);
    }
    wait_async_then_barrier();
    if (!ok) return;
    float rbf = rbfB[l * kP + p];
    for (int k = 0; k < kRBF; ++k) rbf += gsh[sub][k] * Wl[k * kP + p];
    float hj = hbuf[(long)j * kP + p];
    float ep = hj * rbf * f;
    float inv_r = 1.0f / r;
    float s0 = kC1 * vr[e * 3 + 2] * inv_r;   // sph1 = C1*(z,x,y)/r
    float s1 = kC1 * vr[e * 3 + 0] * inv_r;
    float s2 = kC1 * vr[e * 3 + 1] * inv_r;
    atomicAdd(&aggcat[(long)i * (2 * kP) + p], ep);
    atomicAdd(&m_out[(long)i * (3 * kP) + 0 * kP + p], ep * s0);
    atomicAdd(&m_out[(long)i * (3 * kP) + 1 * kP + p], ep * s1);
    atomicAdd(&m_out[(long)i * (3 * kP) + 2 * kP + p], ep * s2);
}

__global__ void cat_norm_k(float* aggcat, const float* m_l) {
    long i = (long)blockIdx.x * blockDim.x + threadIdx.x;
    if (i >= (long)kN * kP) return;
    int n = (int)(i >> 7), p = (int)(i & 127);
    float m0 = m_l[(long)n * 384 + p];
    float m1 = m_l[(long)n * 384 + 128 + p];
    float m2 = m_l[(long)n * 384 + 256 + p];
    aggcat[(long)n * 256 + 128 + p] = sqrtf(m0 * m0 + m1 * m1 + m2 * m2 + 1e-12f);
}

__global__ void bw_m_k(float* grad_m, const float* gradagg, const float* m_l) {
    long i = (long)blockIdx.x * blockDim.x + threadIdx.x;
    if (i >= (long)kN * kP) return;
    int n = (int)(i >> 7), p = (int)(i & 127);
    float m0 = m_l[(long)n * 384 + p];
    float m1 = m_l[(long)n * 384 + 128 + p];
    float m2 = m_l[(long)n * 384 + 256 + p];
    float a1 = sqrtf(m0 * m0 + m1 * m1 + m2 * m2 + 1e-12f);
    float g = gradagg[(long)n * 256 + 128 + p] / a1;
    grad_m[(long)n * 384 + p]       = g * m0;
    grad_m[(long)n * 384 + 128 + p] = g * m1;
    grad_m[(long)n * 384 + 256 + p] = g * m2;
}

// ---------------- fused edge backward ----------------
__global__ __launch_bounds__(256)
void edge_bwd_k(const int* __restrict__ ej, const int* __restrict__ ei, const int* __restrict__ bj,
                const float* __restrict__ image, const float* __restrict__ cell,
                const float* __restrict__ a_l,
                const float* __restrict__ rbfW, const float* __restrict__ rbfB, int l,
                const float* __restrict__ vr, const float* __restrict__ rr,
                const float* __restrict__ fac,
                const float* __restrict__ gradagg, const float* __restrict__ grad_m,
                float* __restrict__ grad_h, float* __restrict__ grad_pos,
                float* __restrict__ grad_strain)
{
    __shared__ float Wl[kRBF * kP];
    __shared__ float gsh[2][kRBF];
    __shared__ float dgsh[2][kRBF];
    __shared__ float red[2][5];
    const int tid = threadIdx.x;
    stage_Wl(Wl, rbfW, l, tid);
    const int sub = tid >> 7, p = tid & 127;
    const long e = (long)blockIdx.x * 2 + sub;
    const bool ok = (e < kE);
    float r = 1.0f, f = 0.0f; int j = 0, i = 0;
    float v0 = 0, v1 = 0, v2 = 0;
    if (ok) {
        r = rr[e]; f = fac[e]; j = ej[e]; i = ei[e];
        v0 = vr[e * 3 + 0]; v1 = vr[e * 3 + 1]; v2 = vr[e * 3 + 2];
    }
    if (ok && p < kRBF) {
        float d = r - kSTEP * (float)p;
        float g = __expf(-kGAMMA * d * d);
        gsh[sub][p] = g;
        dgsh[sub][p] = -2.0f * kGAMMA * d * g;
    }
    if (p < 5) red[sub][p] = 0.0f;
    wait_async_then_barrier();
    float pg[5] = {0, 0, 0, 0, 0};
    if (ok) {
        float rbf = rbfB[l * kP + p], t = 0.0f;
        for (int k = 0; k < kRBF; ++k) {
            rbf += gsh[sub][k] * Wl[k * kP + p];
            t   += dgsh[sub][k] * Wl[k * kP + p];
        }
        float a = a_l[(long)j * kP + p];
        float h = silu(a);
        float inv_r = 1.0f / r;
        float s0 = kC1 * v2 * inv_r, s1 = kC1 * v0 * inv_r, s2 = kC1 * v1 * inv_r;
        float gm0 = grad_m[(long)i * 384 + p];
        float gm1 = grad_m[(long)i * 384 + 128 + p];
        float gm2 = grad_m[(long)i * 384 + 256 + p];
        float ge = gradagg[(long)i * 256 + p] + gm0 * s0 + gm1 * s1 + gm2 * s2;
        atomicAdd(&grad_h[(long)j * kP + p], ge * rbf * f);
        float grbf = ge * h * f;
        float ep = h * rbf * f;
        pg[0] = ge * h * rbf;  // d/dfactor partial
        pg[1] = gm0 * ep;      // d/ds0 partial
        pg[2] = gm1 * ep;
        pg[3] = gm2 * ep;
        pg[4] = grbf * t;      // d/dr via rbf partial
        for (int q = 0; q < 5; ++q) atomicAdd(&red[sub][q], pg[q]);
    }
    __syncthreads();
    if (ok && p == 0) {
        float gfac = red[sub][0], gs0 = red[sub][1], gs1 = red[sub][2], gs2 = red[sub][3];
        float gr = red[sub][4];
        float dfdr = (r < kCUT) ? (-0.5f * kPIf / kCUT) * __sinf(kPIf * r / kCUT) : 0.0f;
        gr += gfac * dfdr;
        float inv_r = 1.0f / r;
        float u = gs0 * v2 + gs1 * v0 + gs2 * v1;
        float c3 = kC1 * u * inv_r * inv_r * inv_r;
        float g0 = kC1 * gs1 * inv_r - c3 * v0 + gr * v0 * inv_r;
        float g1 = kC1 * gs2 * inv_r - c3 * v1 + gr * v1 * inv_r;
        float g2 = kC1 * gs0 * inv_r - c3 * v2 + gr * v2 * inv_r;
        atomicAdd(&grad_pos[(long)j * 3 + 0], g0);
        atomicAdd(&grad_pos[(long)j * 3 + 1], g1);
        atomicAdd(&grad_pos[(long)j * 3 + 2], g2);
        atomicAdd(&grad_pos[(long)i * 3 + 0], -g0);
        atomicAdd(&grad_pos[(long)i * 3 + 1], -g1);
        atomicAdd(&grad_pos[(long)i * 3 + 2], -g2);
        int g = bj[e];
        float gv[3] = {g0, g1, g2};
        for (int b = 0; b < 3; ++b) {
            float sh = image[e * 3 + 0] * cell[g * 9 + 0 + b] +
                       image[e * 3 + 1] * cell[g * 9 + 3 + b] +
                       image[e * 3 + 2] * cell[g * 9 + 6 + b];
            for (int c = 0; c < 3; ++c)
                atomicAdd(&grad_strain[g * 9 + b * 3 + c], sh * gv[c]);
        }
    }
}

// ---------------- head output & backward init ----------------
__global__ void out_k(const float* hD, const float* Wout, const float* bout,
                      const float* E_emb, const int* z, const int* batch, float* energy) {
    int n = blockIdx.x * blockDim.x + threadIdx.x;
    if (n >= kN) return;
    float acc = bout[0] + E_emb[z[n] - 1];
    const float* h = hD + (long)n * kD;
    for (int d = 0; d < kD; ++d) acc += h[d] * Wout[d];
    atomicAdd(&energy[batch[n]], acc);
}
__global__ void head_init_k(float* gA, const float* Wout, const float* a3) {
    long i = (long)blockIdx.x * blockDim.x + threadIdx.x;
    if (i >= (long)kN * kD) return;
    int d = (int)(i & 511);
    gA[i] = Wout[d] * dsilu(a3[i]);
}
__global__ void force_k(float* out_force, const float* grad_pos) {
    long i = (long)blockIdx.x * blockDim.x + threadIdx.x;
    if (i < (long)kN * 3) out_force[i] = -grad_pos[i];
}
__global__ void stress_atoms_k(float* grad_strain, const float* pos,
                               const float* grad_pos, const int* batch) {
    int n = blockIdx.x * blockDim.x + threadIdx.x;
    if (n >= kN) return;
    int g = batch[n];
    for (int b = 0; b < 3; ++b)
        for (int c = 0; c < 3; ++c)
            atomicAdd(&grad_strain[g * 9 + b * 3 + c], pos[n * 3 + b] * grad_pos[n * 3 + c]);
}
__global__ void stress_fin_k(float* out_stress, const float* grad_strain, const float* cell) {
    int g = blockIdx.x * blockDim.x + threadIdx.x;
    if (g >= kG) return;
    const float* a = cell + g * 9;
    float det = a[0] * (a[4] * a[8] - a[5] * a[7])
              - a[1] * (a[3] * a[8] - a[5] * a[6])
              + a[2] * (a[3] * a[7] - a[4] * a[6]);
    float inv = 1.0f / det;
    for (int q = 0; q < 9; ++q) out_stress[g * 9 + q] = grad_strain[g * 9 + q] * inv;
}

// ---------------- host side ----------------
static inline int cdiv(long a, long b) { return (int)((a + b - 1) / b); }

static void launch_gemm(const float* A, int lda, const float* B, long bsk, long bsn,
                        const float* bias, float* C, int ldc, int M, int K, int Nn,
                        hipStream_t s) {
    dim3 g(cdiv(M, BM), cdiv(Nn, BN));
    gemm_bf16_k<<<g, 256, 0, s>>>(A, lda, B, bsk, bsn, bias, C, ldc, M, K, Nn);
}
static void launch_zero(float* p, long n, hipStream_t s) {
    zero_k<<<cdiv(n, 256), 256, 0, s>>>(p, n);
}

extern "C" void kernel_launch(void* const* d_in, const int* in_sizes, int n_in,
                              void* d_out, int out_size, void* d_ws, size_t ws_size,
                              hipStream_t stream) {
    const int*   z      = (const int*)d_in[0];
    const int*   batch  = (const int*)d_in[1];
    const float* frac   = (const float*)d_in[2];
    const float* cell   = (const float*)d_in[3];
    const float* image  = (const float*)d_in[4];
    const int*   ej     = (const int*)d_in[5];
    const int*   ei     = (const int*)d_in[6];
    const int*   bj     = (const int*)d_in[7];

    // ---- resolve param pointers (robust to sorted-key vs insertion-order flattening) ----
    const float *z_emb = nullptr, *E_emb = nullptr, *rbf_W = nullptr, *rbf_b = nullptr;
    const float *W1 = nullptr, *b1 = nullptr, *W2 = nullptr, *b2 = nullptr;
    const float *Wm1 = nullptr, *bm1 = nullptr, *Wm2 = nullptr, *bm2 = nullptr;
    const float *Wm3 = nullptr, *bm3 = nullptr, *Wout = nullptr, *bout = nullptr;
    if (n_in >= 20) {
        const float* wmPtr[3] = {nullptr, nullptr, nullptr}; int wmIdx[3] = {-1, -1, -1}; int nWm = 0;
        const float* p512[4] = {nullptr, nullptr, nullptr, nullptr}; int i512[4] = {-1, -1, -1, -1}; int n512 = 0;
        for (int t = 8; t < n_in; ++t) {
            const float* ptr = (const float*)d_in[t];
            switch (in_sizes[t]) {
                case 53760:  z_emb = ptr; break;
                case 105:    E_emb = ptr; break;
                case 18432:  rbf_W = ptr; break;
                case 196608: W1 = ptr; break;
                case 393216: W2 = ptr; break;
                case 1536:   b2 = ptr; break;
                case 1:      bout = ptr; break;
                case 262144: if (nWm < 3) { wmPtr[nWm] = ptr; wmIdx[nWm] = t; ++nWm; } break;
                case 512:    if (n512 < 4) { p512[n512] = ptr; i512[n512] = t; ++n512; } break;
                case 384:
                    if (t > 8 && in_sizes[t - 1] == 18432) rbf_b = ptr; else b1 = ptr;
                    break;
                default: break;
            }
        }
        Wm1 = wmPtr[0]; Wm2 = wmPtr[1]; Wm3 = wmPtr[2];
        if (n512 == 4) {
            if (i512[0] == wmIdx[0] + 1) { bm1 = p512[0]; bm2 = p512[1]; bm3 = p512[2]; Wout = p512[3]; }
            else                         { Wout = p512[0]; bm1 = p512[1]; bm2 = p512[2]; bm3 = p512[3]; }
        }
    } else {
        // single flat params buffer, JAX sorted-key leaf order
        const float* base = (const float*)d_in[8];
        long o = 0;
        E_emb = base + o; o += 105;
        W1    = base + o; o += 196608;
        W2    = base + o; o += 393216;
        Wm1   = base + o; o += 262144;
        Wm2   = base + o; o += 262144;
        Wm3   = base + o; o += 262144;
        Wout  = base + o; o += 512;
        b1    = base + o; o += 384;
        b2    = base + o; o += 1536;
        bm1   = base + o; o += 512;
        bm2   = base + o; o += 512;
        bm3   = base + o; o += 512;
        bout  = base + o; o += 1;
        rbf_W = base + o; o += 18432;
        rbf_b = base + o; o += 384;
        z_emb = base + o; o += 53760;
    }

    // ---- workspace layout (floats) ----
    float* w = (float*)d_ws;
    size_t off = 0;
    auto alloc = [&](size_t nf) { float* p = w + off; off += nf; return p; };
    float* pos      = alloc((size_t)kN * 3);
    float* vr       = alloc((size_t)kE * 3);
    float* rr       = alloc((size_t)kE);
    float* fac      = alloc((size_t)kE);
    float* x_save   = alloc((size_t)4 * kN * kD);
    float* a_save   = alloc((size_t)kL * kN * kP);
    float* m_save   = alloc((size_t)kL * kN * 3 * kP);
    float* hbuf     = alloc((size_t)kN * kP);
    float* aggcat   = alloc((size_t)kN * 2 * kP);
    float* dxbuf    = alloc((size_t)kN * kD);
    float* mlpa     = alloc((size_t)3 * kN * kD);
    float* hD       = alloc((size_t)kN * kD);
    float* gA       = alloc((size_t)kN * kD);
    float* gB       = alloc((size_t)kN * kD);
    float* grad_x   = alloc((size_t)kN * kD);
    float* gradagg  = alloc((size_t)kN * 2 * kP);
    float* grad_m   = alloc((size_t)kN * 3 * kP);
    float* grad_h   = alloc((size_t)kN * kP);
    float* grad_pos = alloc((size_t)kN * 3);
    float* grad_str = alloc((size_t)kG * 9);

    float* out_energy = (float*)d_out;
    float* out_force  = out_energy + kG;
    float* out_stress = out_force + (long)kN * 3;

    const long ND = (long)kN * kD, NP = (long)kN * kP;
    const int edgeBlocks = cdiv(kE, 2);

    // ---- forward geometry ----
    pos_k<<<cdiv(kN, 256), 256, 0, stream>>>(pos, frac, cell, batch);
    edge_geom_k<<<cdiv(kE, 256), 256, 0, stream>>>(pos, cell, image, ej, ei, bj, vr, rr, fac);

    // ---- embedding + message passing layers ----
    embed_k<<<cdiv(ND, 256), 256, 0, stream>>>(x_save, z_emb, z);
    launch_zero(m_save, (size_t)kL * kN * 3 * kP, stream);
    for (int l = 0; l < kL; ++l) {
        float* xl  = x_save + (size_t)l * ND;
        float* xl1 = x_save + (size_t)(l + 1) * ND;
        float* al  = a_save + (size_t)l * NP;
        float* ml  = m_save + (size_t)l * kN * 3 * kP;
        launch_gemm(xl, kD, W1 + (size_t)l * kD * kP, kP, 1, b1 + l * kP,
                    al, kP, kN, kD, kP, stream);
        silu_k<<<cdiv(NP, 256), 256, 0, stream>>>(hbuf, al, NP);
        launch_zero(aggcat, (size_t)kN * 2 * kP, stream);
        edge_fwd_k<<<edgeBlocks, 256, 0, stream>>>(ej, ei, hbuf, rbf_W, rbf_b, l,
                                                   vr, rr, fac, aggcat, ml);
        cat_norm_k<<<cdiv(NP, 256), 256, 0, stream>>>(aggcat, ml);
        launch_gemm(aggcat, 2 * kP, W2 + (size_t)l * 2 * kP * kD, kD, 1, b2 + l * kD,
                    dxbuf, kD, kN, 2 * kP, kD, stream);
        add3_k<<<cdiv(ND, 256), 256, 0, stream>>>(xl1, xl, dxbuf, ND);
    }

    // ---- MLP head ----
    float* x3 = x_save + (size_t)3 * ND;
    launch_gemm(x3, kD, Wm1, kD, 1, bm1, mlpa + 0 * ND, kD, kN, kD, kD, stream);
    silu_k<<<cdiv(ND, 256), 256, 0, stream>>>(hD, mlpa + 0 * ND, ND);
    launch_gemm(hD, kD, Wm2, kD, 1, bm2, mlpa + 1 * ND, kD, kN, kD, kD, stream);
    silu_k<<<cdiv(ND, 256), 256, 0, stream>>>(hD, mlpa + 1 * ND, ND);
    launch_gemm(hD, kD, Wm3, kD, 1, bm3, mlpa + 2 * ND, kD, kN, kD, kD, stream);
    silu_k<<<cdiv(ND, 256), 256, 0, stream>>>(hD, mlpa + 2 * ND, ND);
    launch_zero(out_energy, kG, stream);
    out_k<<<cdiv(kN, 256), 256, 0, stream>>>(hD, Wout, bout, E_emb, z, batch, out_energy);

    // ---- backward: MLP head (transposed-B GEMMs) ----
    head_init_k<<<cdiv(ND, 256), 256, 0, stream>>>(gA, Wout, mlpa + 2 * ND);
    launch_gemm(gA, kD, Wm3, 1, kD, nullptr, gB, kD, kN, kD, kD, stream);
    muldsilu_k<<<cdiv(ND, 256), 256, 0, stream>>>(gB, mlpa + 1 * ND, ND);
    launch_gemm(gB, kD, Wm2, 1, kD, nullptr, gA, kD, kN, kD, kD, stream);
    muldsilu_k<<<cdiv(ND, 256), 256, 0, stream>>>(gA, mlpa + 0 * ND, ND);
    launch_gemm(gA, kD, Wm1, 1, kD, nullptr, grad_x, kD, kN, kD, kD, stream);

    // ---- backward: layers (reverse) ----
    launch_zero(grad_pos, (size_t)kN * 3, stream);
    launch_zero(grad_str, kG * 9, stream);
    for (int l = kL - 1; l >= 0; --l) {
        float* al = a_save + (size_t)l * NP;
        float* ml = m_save + (size_t)l * kN * 3 * kP;
        launch_gemm(grad_x, kD, W2 + (size_t)l * 2 * kP * kD, 1, kD, nullptr,
                    gradagg, 2 * kP, kN, kD, 2 * kP, stream);
        bw_m_k<<<cdiv(NP, 256), 256, 0, stream>>>(grad_m, gradagg, ml);
        launch_zero(grad_h, NP, stream);
        edge_bwd_k<<<edgeBlocks, 256, 0, stream>>>(ej, ei, bj, image, cell, al,
                                                   rbf_W, rbf_b, l, vr, rr, fac,
                                                   gradagg, grad_m, grad_h,
                                                   grad_pos, grad_str);
        muldsilu_k<<<cdiv(NP, 256), 256, 0, stream>>>(grad_h, al, NP);
        launch_gemm(grad_h, kP, W1 + (size_t)l * kD * kP, 1, kP, nullptr,
                    dxbuf, kD, kN, kP, kD, stream);
        addip_k<<<cdiv(ND, 256), 256, 0, stream>>>(grad_x, dxbuf, ND);
    }

    // ---- outputs: force & stress ----
    force_k<<<cdiv((long)kN * 3, 256), 256, 0, stream>>>(out_force, grad_pos);
    stress_atoms_k<<<cdiv(kN, 256), 256, 0, stream>>>(grad_str, pos, grad_pos, batch);
    stress_fin_k<<<1, 64, 0, stream>>>(out_stress, grad_str, cell);

    (void)in_sizes; (void)out_size; (void)ws_size;
}
